// PointCapsNet_3521873183195
// MI455X (gfx1250) — matrix operand order repaired and checked by reading, not compiled, not run on hardware
//
#include <hip/hip_runtime.h>
#include <math.h>

typedef __attribute__((ext_vector_type(16))) _Float16 v16h;
typedef __attribute__((ext_vector_type(8)))  float    v8f;
typedef __attribute__((ext_vector_type(4)))  unsigned u32x4;
typedef __attribute__((ext_vector_type(8)))  int      i32x8;
typedef __attribute__((ext_vector_type(4)))  int      i32x4;

#if defined(__gfx1250__) && __has_builtin(__builtin_amdgcn_tensor_load_to_lds)
#define HAVE_TDM 1
#else
#define HAVE_TDM 0
#endif

#define BB 4
#define NN 2048
#define PC 1024
#define PV 16
#define LL 64
#define DVV 64
#define NPRIM 32
#define BOT 66
#define COLS (BB*NN)     /* 8192 */
#define ROWS (PV*PC)     /* 16384 */
#define EPS_BN 1e-5f

// ---------------- helpers ----------------
__device__ __forceinline__ unsigned fmap(float f) {
  unsigned u = __float_as_uint(f);
  return (u & 0x80000000u) ? ~u : (u | 0x80000000u);
}
__device__ __forceinline__ float funmap(unsigned m) {
  unsigned u = (m & 0x80000000u) ? (m & 0x7FFFFFFFu) : ~m;
  return __uint_as_float(u);
}

// ---------------- zero ----------------
__global__ void k_zero(float* p, int n) {
  int i = blockIdx.x * blockDim.x + threadIdx.x;
  if (i < n) p[i] = 0.0f;
}

// ---------------- conv1 raw (B,3,N)x(64,3) -> (B,64,N) ----------------
__global__ void k_conv1(const float* __restrict__ x, const float* __restrict__ w,
                        const float* __restrict__ b, float* __restrict__ h1raw) {
  int idx = blockIdx.x * blockDim.x + threadIdx.x; // (b,o,n)
  if (idx >= BB * 64 * NN) return;
  int n = idx % NN; int rem = idx / NN; int o = rem % 64; int bb = rem / 64;
  float acc = b[o];
  #pragma unroll
  for (int c = 0; c < 3; ++c)
    acc += x[(bb * 3 + c) * NN + n] * w[o * 3 + c];
  h1raw[idx] = acc;
}

// ---------------- per-channel stats over (B,N) ----------------
__global__ void k_chanstats(const float* __restrict__ d, int C,
                            float* __restrict__ sum, float* __restrict__ sq) {
  __shared__ float rs[256], rq[256];
  int c = blockIdx.x, t = threadIdx.x;
  float s = 0.f, q = 0.f;
  for (int i = t; i < BB * NN; i += 256) {
    int bb = i / NN, n = i % NN;
    float v = d[((size_t)bb * C + c) * NN + n];
    s += v; q += v * v;
  }
  rs[t] = s; rq[t] = q; __syncthreads();
  for (int st = 128; st > 0; st >>= 1) {
    if (t < st) { rs[t] += rs[t + st]; rq[t] += rq[t + st]; }
    __syncthreads();
  }
  if (t == 0) { sum[c] = rs[0]; sq[c] = rq[0]; }
}

__global__ void k_bnfin(const float* __restrict__ sum, const float* __restrict__ sq,
                        const float* __restrict__ g, const float* __restrict__ be,
                        float* __restrict__ sc, float* __restrict__ sh,
                        int C, float invc) {
  int t = blockIdx.x * blockDim.x + threadIdx.x;
  if (t >= C) return;
  float mean = sum[t] * invc;
  float var  = sq[t] * invc - mean * mean;
  float rstd = rsqrtf(var + EPS_BN);
  float gg = g[t];
  sc[t] = gg * rstd;
  sh[t] = be[t] - gg * rstd * mean;
}

// ---------------- conv2: bn+relu(h1) @ (128,64) -> h2raw ----------------
__global__ void k_conv2(const float* __restrict__ h1raw, const float* __restrict__ sc1,
                        const float* __restrict__ sh1, const float* __restrict__ w,
                        const float* __restrict__ b, float* __restrict__ h2raw) {
  __shared__ float h1n[64][128];
  int n0 = blockIdx.x * 128, bb = blockIdx.y, t = threadIdx.x;
  for (int i = t; i < 64 * 128; i += 256) {
    int c = i >> 7, nn = i & 127;
    float raw = h1raw[((size_t)bb * 64 + c) * NN + n0 + nn];
    h1n[c][nn] = fmaxf(sc1[c] * raw + sh1[c], 0.0f);
  }
  __syncthreads();
  int nn = t & 127, og = t >> 7;
  for (int o = og; o < 128; o += 2) {
    float acc = b[o];
    const float* wr = w + o * 64;
    #pragma unroll 16
    for (int c = 0; c < 64; ++c) acc += h1n[c][nn] * wr[c];
    h2raw[((size_t)bb * 128 + o) * NN + n0 + nn] = acc;
  }
}

// ---------------- pack bn+relu(h2) to f16, TRANSPOSED: [col=b*N+n][K=128] ----
// LDS tile transpose keeps both global sides coalesced; the transposed layout
// lets the GEMM load a whole 16-half B fragment as one aligned 32B vector.
__global__ void k_pack_h2(const float* __restrict__ h2raw, const float* __restrict__ sc2,
                          const float* __restrict__ sh2, _Float16* __restrict__ h2fT) {
  __shared__ _Float16 tile[128 * 64];
  int n0 = blockIdx.x * 64, bb = blockIdx.y, t = threadIdx.x;
  for (int i = t; i < 128 * 64; i += 256) {
    int c = i >> 6, nn = i & 63;
    float v = fmaxf(sc2[c] * h2raw[((size_t)bb * 128 + c) * NN + n0 + nn] + sh2[c], 0.0f);
    tile[c * 64 + nn] = (_Float16)v;
  }
  __syncthreads();
  for (int i = t; i < 64 * 128; i += 256) {
    int nn = i >> 7, c = i & 127;
    h2fT[((size_t)bb * NN + n0 + nn) * 128 + c] = tile[c * 64 + nn];
  }
}

__global__ void k_pack_pw(const float* __restrict__ pw, _Float16* __restrict__ pwf) {
  int i = blockIdx.x * blockDim.x + threadIdx.x;
  if (i < ROWS * 128) pwf[i] = (_Float16)pw[i];
}

// ---------------- WMMA GEMM + fused row stats / per-b max ----------------
// C[row,col] = sum_k pwf[row,k]*h2fT[col,k] (+p_b[row]); rows=16384, cols=8192, K=128
__global__ __launch_bounds__(256) void k_gemm(
    const _Float16* __restrict__ pwf, const _Float16* __restrict__ h2fT,
    const float* __restrict__ pb,
    float* __restrict__ rsum, float* __restrict__ rsq, unsigned* __restrict__ rmax) {
  __shared__ __align__(16) _Float16 Ald[16 * 128];
  __shared__ float pbl[16], lsum[16], lsq[16];
  __shared__ unsigned lmax[16];
  int t = threadIdx.x;
  int row0 = blockIdx.y * 16;

  // stage A tile (16 rows x 128 K = 4KB contiguous) via Tensor Data Mover
  if (t < 32) {
#if HAVE_TDM
    unsigned long long ga = (unsigned long long)(const void*)(pwf + (size_t)row0 * 128);
    unsigned ldsa = (unsigned)(unsigned long long)(const void*)&Ald[0]; // flat[31:0]==LDS offset
    u32x4 g0;
    g0[0] = 1u;                                   // count=1 valid descriptor
    g0[1] = ldsa;                                 // lds_addr
    g0[2] = (unsigned)ga;                         // global_addr[31:0]
    g0[3] = (unsigned)((ga >> 32) & 0x1FFFFFFull) // global_addr[56:32]
          | (2u << 30);                           // type=2 ("image")
    i32x8 g1;
    g1[0] = (int)(1u << 16);                      // data_size=1 (2 bytes)
    g1[1] = (int)(2048u << 16);                   // tensor_dim0 = 2048
    g1[2] = (int)(1u << 16);                      // tensor_dim1 = 1
    g1[3] = (int)(2048u << 16);                   // tile_dim0 = 2048
    g1[4] = 0;                                    // tile_dim1/2 unused
    g1[5] = 2048;                                 // tensor_dim0_stride
    g1[6] = 0; g1[7] = 0;
    i32x4 gz; gz[0] = 0; gz[1] = 0; gz[2] = 0; gz[3] = 0;
    i32x8 gz8; gz8[0] = 0; gz8[1] = 0; gz8[2] = 0; gz8[3] = 0;
    gz8[4] = 0; gz8[5] = 0; gz8[6] = 0; gz8[7] = 0;
    __builtin_amdgcn_tensor_load_to_lds(g0, g1, gz, gz, gz8, 0);
    __builtin_amdgcn_s_wait_tensorcnt(0);
#else
    for (int i = t; i < 256; i += 32)
      ((uint4*)Ald)[i] = ((const uint4*)(pwf + (size_t)row0 * 128))[i];
#endif
  }
  if (t < 16) { pbl[t] = pb[row0 + t]; lsum[t] = 0.f; lsq[t] = 0.f; lmax[t] = 0u; }
  __syncthreads();

  int wave = t >> 5, lane = t & 31, hl = lane & 15, hi = lane >> 4;
  int col = blockIdx.x * 128 + wave * 16 + hl;

  v8f acc = {};
  #pragma unroll
  for (int kt = 0; kt < 4; ++kt) {
    v16h a;
    int a0 = kt * 32 + hi * 16, a1 = kt * 32 + hi * 16 + 8;
    #pragma unroll
    for (int j = 0; j < 8; ++j) {
      a[j]     = Ald[hl * 128 + a0 + j];
      a[8 + j] = Ald[hl * 128 + a1 + j];
    }
    // transposed layout: 16 consecutive K-halfs per lane -> one 32B vector load
    v16h bf = *(const v16h*)(h2fT + (size_t)col * 128 + kt * 32 + hi * 16);
    acc = __builtin_amdgcn_wmma_f32_16x16x32_f16(false, a, false, bf,
                                                 (short)0, acc, false, false);
  }
  // epilogue: per-row (sum, sumsq, max) over this wave's 16 cols
  #pragma unroll
  for (int r = 0; r < 8; ++r) {
    int m = hi * 8 + r;
    float val = acc[r] + pbl[m];
    float s = val, q = val * val, mx = val;
    #pragma unroll
    for (int off = 8; off > 0; off >>= 1) {
      s  += __shfl_xor(s, off, 16);
      q  += __shfl_xor(q, off, 16);
      mx  = fmaxf(mx, __shfl_xor(mx, off, 16));
    }
    if (hl == 0) {
      atomicAdd(&lsum[m], s);
      atomicAdd(&lsq[m], q);
      atomicMax(&lmax[m], fmap(mx));
    }
  }
  __syncthreads();
  if (t < 16) {
    int row = row0 + t;
    int bb = blockIdx.x >> 4; // 128-col tile lies within one batch slab
    atomicAdd(&rsum[row], lsum[t]);
    atomicAdd(&rsq[row], lsq[t]);
    atomicMax(&rmax[(size_t)bb * ROWS + row], lmax[t]);
  }
}

// ---------------- finalize u = bn(max) -> (B,PC,PV) ----------------
__global__ void k_ufinal(const float* __restrict__ rsum, const float* __restrict__ rsq,
                         const unsigned* __restrict__ rmax, const float* __restrict__ pg,
                         const float* __restrict__ pbe, float* __restrict__ ut) {
  int idx = blockIdx.x * blockDim.x + threadIdx.x;
  if (idx >= BB * ROWS) return;
  int bb = idx / ROWS, row = idx % ROWS;
  float invc = 1.0f / (float)COLS;
  float mean = rsum[row] * invc;
  float var  = rsq[row] * invc - mean * mean;
  float rstd = rsqrtf(var + EPS_BN);
  float mx = funmap(rmax[idx]);
  float v = pg[row] * (mx - mean) * rstd + pbe[row];
  int k = row / PC, o = row % PC;
  ut[((size_t)bb * PC + o) * PV + k] = v;
}

// ---------------- squash_prim over PV ----------------
__global__ void k_squash(float* __restrict__ ut, float* __restrict__ unorm) {
  int i = blockIdx.x * blockDim.x + threadIdx.x; // (b,p)
  if (i >= BB * PC) return;
  float* u = ut + (size_t)i * PV;
  float sq = 0.f;
  #pragma unroll
  for (int d = 0; d < PV; ++d) sq += u[d] * u[d];
  float f = sq / ((1.0f + sq) * sqrtf(sq));
  #pragma unroll
  for (int d = 0; d < PV; ++d) u[d] *= f;
  unorm[i] = sq / (1.0f + sq);
}

// ---------------- u_hat = W @ u, norm-clamped ----------------
__global__ void k_uhat(const float* __restrict__ W, const float* __restrict__ ut,
                       const float* __restrict__ unorm, float* __restrict__ uhat) {
  __shared__ float ul[BB][PV], un[BB], red[64], snorm[BB];
  int p = blockIdx.x, l = blockIdx.y, t = threadIdx.x;
  {
    int bb = t >> 4, d = t & 15;
    ul[bb][d] = ut[((size_t)bb * PC + p) * PV + d];
  }
  if (t < BB) un[t] = unorm[(size_t)t * PC + p];
  __syncthreads();

  const float* wp = W + ((((size_t)l * PC + p) * DVV) + t) * PV;
  __builtin_prefetch(wp + DVV * PV, 0, 1);
  float4 w0 = ((const float4*)wp)[0], w1 = ((const float4*)wp)[1];
  float4 w2 = ((const float4*)wp)[2], w3 = ((const float4*)wp)[3];
  float uh[BB];
  #pragma unroll
  for (int bb = 0; bb < BB; ++bb) {
    const float* uu = ul[bb];
    uh[bb] = w0.x*uu[0] + w0.y*uu[1] + w0.z*uu[2] + w0.w*uu[3]
           + w1.x*uu[4] + w1.y*uu[5] + w1.z*uu[6] + w1.w*uu[7]
           + w2.x*uu[8] + w2.y*uu[9] + w2.z*uu[10]+ w2.w*uu[11]
           + w3.x*uu[12]+ w3.y*uu[13]+ w3.z*uu[14]+ w3.w*uu[15];
  }
  #pragma unroll
  for (int bb = 0; bb < BB; ++bb) {
    red[t] = uh[bb] * uh[bb];
    __syncthreads();
    for (int s = 32; s > 0; s >>= 1) { if (t < s) red[t] += red[t + s]; __syncthreads(); }
    if (t == 0) {
      float nrm = sqrtf(red[0]);
      snorm[bb] = fminf(nrm, un[bb]) / nrm;
    }
    __syncthreads();
  }
  size_t base = (((size_t)0 + l) * PC + p) * DVV + t;
  #pragma unroll
  for (int bb = 0; bb < BB; ++bb)
    uhat[(size_t)bb * LL * PC * DVV + base] = uh[bb] * snorm[bb];
}

// ---------------- routing: softmax over L ----------------
__global__ void k_softmax(const float* __restrict__ blog, float* __restrict__ c) {
  int i = blockIdx.x * blockDim.x + threadIdx.x; // (b,p)
  if (i >= BB * PC) return;
  int bb = i / PC, p = i % PC;
  float mx = -1e30f;
  for (int l = 0; l < LL; ++l)
    mx = fmaxf(mx, blog[((size_t)bb * LL + l) * PC + p]);
  float s = 0.f;
  for (int l = 0; l < LL; ++l)
    s += __expf(blog[((size_t)bb * LL + l) * PC + p] - mx);
  float inv = 1.0f / s;
  for (int l = 0; l < LL; ++l) {
    size_t idx = ((size_t)bb * LL + l) * PC + p;
    c[idx] = __expf(blog[idx] - mx) * inv;
  }
}

// ---------------- s = sum_p c*u_hat + 0.01 ; v = squash_gamma(s) ----------------
__global__ void k_sv(const float* __restrict__ c, const float* __restrict__ uhat,
                     float* __restrict__ vbuf) {
  __shared__ float sacc[DVV], red[64];
  int bl = blockIdx.x, t = threadIdx.x; // bl = b*L + l
  if (t < DVV) sacc[t] = 0.f;
  __syncthreads();
  float4 acc[16];
  #pragma unroll
  for (int j = 0; j < 16; ++j) acc[j] = make_float4(0.f, 0.f, 0.f, 0.f);
  for (int p = t; p < PC; p += 256) {
    float cv = c[(size_t)bl * PC + p];
    const float4* up = (const float4*)(uhat + ((size_t)bl * PC + p) * DVV);
    #pragma unroll
    for (int j = 0; j < 16; ++j) {
      float4 q = up[j];
      acc[j].x += cv * q.x; acc[j].y += cv * q.y;
      acc[j].z += cv * q.z; acc[j].w += cv * q.w;
    }
  }
  #pragma unroll
  for (int j = 0; j < 16; ++j) {
    atomicAdd(&sacc[4 * j + 0], acc[j].x);
    atomicAdd(&sacc[4 * j + 1], acc[j].y);
    atomicAdd(&sacc[4 * j + 2], acc[j].z);
    atomicAdd(&sacc[4 * j + 3], acc[j].w);
  }
  __syncthreads();
  if (t < DVV) {
    float s = sacc[t] + 0.01f;
    red[t] = s * s;
    __syncthreads();
    for (int st = 32; st > 0; st >>= 1) { if (t < st) red[t] += red[t + st]; __syncthreads(); }
    float sq = red[0] + 1e-12f;
    float vo = sq / (1.0f + sq) * (s * rsqrtf(sq)) + 1e-12f;
    vbuf[(size_t)bl * DVV + t] = vo;
  }
}

// ---------------- d = ||v - u_hat||, accumulate global mean ----------------
__global__ void k_dist(const float* __restrict__ vbuf, const float* __restrict__ uhat,
                       float* __restrict__ d, float* __restrict__ dsum) {
  __shared__ float vv[DVV], red[256];
  int bl = blockIdx.x, t = threadIdx.x;
  if (t < DVV) vv[t] = vbuf[(size_t)bl * DVV + t];
  __syncthreads();
  float loc = 0.f;
  for (int p = t; p < PC; p += 256) {
    const float4* up = (const float4*)(uhat + ((size_t)bl * PC + p) * DVV);
    float s = 0.f;
    #pragma unroll
    for (int j = 0; j < 16; ++j) {
      float4 q = up[j];
      float d0 = vv[4*j+0] - q.x, d1 = vv[4*j+1] - q.y;
      float d2 = vv[4*j+2] - q.z, d3 = vv[4*j+3] - q.w;
      s += d0*d0 + d1*d1 + d2*d2 + d3*d3;
    }
    float dv = sqrtf(s);
    d[(size_t)bl * PC + p] = dv;
    loc += dv;
  }
  red[t] = loc; __syncthreads();
  for (int st = 128; st > 0; st >>= 1) { if (t < st) red[t] += red[t + st]; __syncthreads(); }
  if (t == 0) atomicAdd(dsum, red[0]);
}

__global__ void k_blog(const float* __restrict__ d, const float* __restrict__ dsum,
                       float* __restrict__ blog) {
  int i = blockIdx.x * blockDim.x + threadIdx.x;
  if (i >= BB * LL * PC) return;
  float d_o = dsum[0] * (1.0f / (float)(BB * LL * PC));
  float t_num = logf(0.9f * (LL - 1)) - logf(0.1f);
  float tt = t_num / (0.5f * d_o - d_o + 1e-12f);
  blog[i] = tt * d[i];
}

// ---------------- T scalar ----------------
__global__ void k_T(const float* __restrict__ c, float* __restrict__ out) {
  __shared__ float red[256];
  int t = threadIdx.x;
  float acc = 0.f;
  float invlog = 1.0f / logf((float)LL);
  for (int i = t; i < BB * PC; i += 256) {
    int bb = i / PC, p = i % PC;
    float ent = 0.f;
    for (int l = 0; l < LL; ++l) {
      float cv = c[((size_t)bb * LL + l) * PC + p];
      ent -= cv * logf(cv + 1e-12f);
    }
    acc += 1.0f - ent * invlog;
  }
  red[t] = acc; __syncthreads();
  for (int st = 128; st > 0; st >>= 1) { if (t < st) red[t] += red[t + st]; __syncthreads(); }
  if (t == 0) out[0] = red[0] / (float)(BB * PC);
}

// ---------------- copy v_j + D scalar ----------------
__global__ void k_vjD(const float* __restrict__ vbuf, float* __restrict__ vj_out,
                      float* __restrict__ D_out) {
  __shared__ float red[64];
  int t = threadIdx.x;
  for (int i = t; i < BB * LL * DVV; i += 256) vj_out[i] = vbuf[i];
  if (t < LL) {
    float nb[BB];
    float mean = 0.f;
    #pragma unroll
    for (int bb = 0; bb < BB; ++bb) {
      float s = 0.f;
      for (int v = 0; v < DVV; ++v) {
        float x = vbuf[((size_t)bb * LL + t) * DVV + v];
        s += x * x;
      }
      nb[bb] = sqrtf(s);
      mean += nb[bb];
    }
    mean *= 0.25f;
    float var = 0.f;
    #pragma unroll
    for (int bb = 0; bb < BB; ++bb) { float dd = nb[bb] - mean; var += dd * dd; }
    red[t] = sqrtf(var * 0.25f);
  }
  __syncthreads();
  if (t < 32) { if (t + 32 < 64) red[t] = fmaxf(red[t], red[t + 32]); }
  __syncthreads();
  for (int st = 16; st > 0; st >>= 1) {
    if (t < st) red[t] = fmaxf(red[t], red[t + st]);
    __syncthreads();
  }
  if (t == 0) D_out[0] = red[0];
}

// ---------------- decoder ----------------
__device__ void dec_layer(const float* __restrict__ w, const float* __restrict__ bb,
                          const float* __restrict__ g, const float* __restrict__ be,
                          int Cin, int Cout, int pr,
                          const float* bin, float* bout, float* st) {
  int t = threadIdx.x, col = t;
  for (int o = 0; o < Cout; ++o) {
    float acc = bb[pr * Cout + o];
    const float* wr = w + ((size_t)pr * Cout + o) * Cin;
    for (int cch = 0; cch < Cin; ++cch) acc += bin[cch * 256 + col] * wr[cch];
    bout[o * 256 + col] = acc;
  }
  __syncthreads();
  float* sc = st; float* sh = st + 128;
  if (t < Cout) {
    float s = 0.f, q = 0.f;
    for (int i = 0; i < 256; ++i) {
      int cc = (t + i) & 255;           // staggered -> conflict-free
      float v = bout[t * 256 + cc];
      s += v; q += v * v;
    }
    float mean = s * (1.0f / 256.0f);
    float var  = q * (1.0f / 256.0f) - mean * mean;
    float rstd = rsqrtf(var + EPS_BN);
    float gg = g[pr * Cout + t];
    sc[t] = gg * rstd;
    sh[t] = be[pr * Cout + t] - gg * rstd * mean;
  }
  __syncthreads();
  for (int i = t; i < Cout * 256; i += 256) {
    int o = i >> 8;
    bout[i] = fmaxf(sc[o] * bout[i] + sh[o], 0.0f);
  }
  __syncthreads();
}

__global__ void k_decoder(const float* __restrict__ rg, const float* __restrict__ vbuf,
    const float* __restrict__ d1w, const float* __restrict__ d1b,
    const float* __restrict__ d1g, const float* __restrict__ d1be,
    const float* __restrict__ d2w, const float* __restrict__ d2b,
    const float* __restrict__ d2g, const float* __restrict__ d2be,
    const float* __restrict__ d3w, const float* __restrict__ d3b,
    const float* __restrict__ d3g, const float* __restrict__ d3be,
    const float* __restrict__ d4w, const float* __restrict__ d4b,
    float* __restrict__ recon) {
  extern __shared__ float smem[];
  float* buf0 = smem;
  float* buf1 = smem + BOT * 256;
  float* st   = smem + 2 * BOT * 256;
  int pr = blockIdx.x, t = threadIdx.x;
  // build y: ch 0..1 = rand_grids, 2..65 = lat = v_j[b][l][ch-2]
  for (int i = t; i < BOT * 256; i += 256) {
    int ch = i / 256, col = i % 256, bb = col >> 6, l = col & 63;
    float v;
    if (ch < 2) v = rg[(((size_t)pr * BB + bb) * 2 + ch) * LL + l];
    else        v = vbuf[((size_t)bb * LL + l) * DVV + (ch - 2)];
    buf0[i] = v;
  }
  __syncthreads();
  dec_layer(d1w, d1b, d1g, d1be, BOT, BOT,      pr, buf0, buf1, st);
  dec_layer(d2w, d2b, d2g, d2be, BOT, BOT / 2,  pr, buf1, buf0, st);
  dec_layer(d3w, d3b, d3g, d3be, BOT / 2, BOT / 4, pr, buf0, buf1, st);
  // layer 4: tanh(conv), write recon[b][ch][pr*64+l]
  int col = t, bb = col >> 6, l = col & 63;
  for (int o = 0; o < 3; ++o) {
    float acc = d4b[pr * 3 + o];
    const float* wr = d4w + ((size_t)pr * 3 + o) * (BOT / 4);
    #pragma unroll
    for (int cch = 0; cch < BOT / 4; ++cch) acc += buf1[cch * 256 + col] * wr[cch];
    recon[((size_t)bb * 3 + o) * (NPRIM * LL) + pr * LL + l] = tanhf(acc);
  }
}

// ================= host launch =================
extern "C" void kernel_launch(void* const* d_in, const int* in_sizes, int n_in,
                              void* d_out, int out_size, void* d_ws, size_t ws_size,
                              hipStream_t stream) {
  const float* x    = (const float*)d_in[0];
  const float* rg   = (const float*)d_in[1];
  const float* c1w  = (const float*)d_in[2];
  const float* c1b  = (const float*)d_in[3];
  const float* bn1g = (const float*)d_in[4];
  const float* bn1b = (const float*)d_in[5];
  const float* c2w  = (const float*)d_in[6];
  const float* c2b  = (const float*)d_in[7];
  const float* bn2g = (const float*)d_in[8];
  const float* bn2b = (const float*)d_in[9];
  const float* pw   = (const float*)d_in[10];
  const float* pb   = (const float*)d_in[11];
  const float* pg   = (const float*)d_in[12];
  const float* pbe  = (const float*)d_in[13];
  const float* W    = (const float*)d_in[14];
  const float* d1w  = (const float*)d_in[15];
  const float* d1b  = (const float*)d_in[16];
  const float* d1g  = (const float*)d_in[17];
  const float* d1be = (const float*)d_in[18];
  const float* d2w  = (const float*)d_in[19];
  const float* d2b  = (const float*)d_in[20];
  const float* d2g  = (const float*)d_in[21];
  const float* d2be = (const float*)d_in[22];
  const float* d3w  = (const float*)d_in[23];
  const float* d3b  = (const float*)d_in[24];
  const float* d3g  = (const float*)d_in[25];
  const float* d3be = (const float*)d_in[26];
  const float* d4w  = (const float*)d_in[27];
  const float* d4b  = (const float*)d_in[28];

  float* ws = (float*)d_ws;
  // workspace layout (float offsets)
  float*    h1raw  = ws;                              // 524288
  float*    h2raw  = ws + 524288;                     // 1048576
  float*    st1s   = ws + 1572864;                    // 64
  float*    st1q   = st1s + 64;
  float*    sc1    = st1s + 128;
  float*    sh1    = st1s + 192;
  float*    st2s   = ws + 1573120;                    // 128
  float*    st2q   = st2s + 128;
  float*    sc2    = st2s + 256;
  float*    sh2    = st2s + 384;
  _Float16* h2fT   = (_Float16*)(ws + 1573632);       // 1048576 halfs [col][K]
  _Float16* pwf    = (_Float16*)(ws + 2097920);       // 2097152 halfs [row][K]
  float*    rsum   = ws + 3146496;                    // 16384
  float*    rsq    = rsum + ROWS;
  unsigned* rmax   = (unsigned*)(ws + 3179264);       // 65536
  float*    ut     = ws + 3244800;                    // 65536
  float*    unorm  = ws + 3310336;                    // 4096
  float*    uhat   = ws + 3314432;                    // 16777216
  float*    blog   = ws + 20091648;                   // 262144
  float*    cbuf   = ws + 20353792;                   // 262144
  float*    dbuf   = ws + 20615936;                   // 262144
  float*    vbuf   = ws + 20878080;                   // 16384
  float*    dsum   = ws + 20894464;                   // 1

  float* out  = (float*)d_out;
  float* vj_o = out;            // 16384
  float* rc_o = out + 16384;    // 24576
  float* T_o  = out + 40960;
  float* D_o  = out + 40961;

  // zero atomic-accumulated regions
  k_zero<<<(2 * ROWS + 255) / 256, 256, 0, stream>>>(rsum, 2 * ROWS);
  k_zero<<<(BB * ROWS + 255) / 256, 256, 0, stream>>>((float*)rmax, BB * ROWS);
  k_zero<<<(BB * LL * PC + 255) / 256, 256, 0, stream>>>(blog, BB * LL * PC);

  // encoder
  k_conv1<<<(BB * 64 * NN + 255) / 256, 256, 0, stream>>>(x, c1w, c1b, h1raw);
  k_chanstats<<<64, 256, 0, stream>>>(h1raw, 64, st1s, st1q);
  k_bnfin<<<1, 64, 0, stream>>>(st1s, st1q, bn1g, bn1b, sc1, sh1, 64, 1.0f / (BB * NN));
  k_conv2<<<dim3(NN / 128, BB), 256, 0, stream>>>(h1raw, sc1, sh1, c2w, c2b, h2raw);
  k_chanstats<<<128, 256, 0, stream>>>(h2raw, 128, st2s, st2q);
  k_bnfin<<<1, 128, 0, stream>>>(st2s, st2q, bn2g, bn2b, sc2, sh2, 128, 1.0f / (BB * NN));
  k_pack_h2<<<dim3(NN / 64, BB), 256, 0, stream>>>(h2raw, sc2, sh2, h2fT);
  k_pack_pw<<<(ROWS * 128 + 255) / 256, 256, 0, stream>>>(pw, pwf);

  // primary caps GEMM (WMMA + TDM A-tile) with fused bn-stats + per-batch max
  k_gemm<<<dim3(COLS / 128, ROWS / 16), 256, 0, stream>>>(pwf, h2fT, pb, rsum, rsq, rmax);
  k_ufinal<<<(BB * ROWS + 255) / 256, 256, 0, stream>>>(rsum, rsq, rmax, pg, pbe, ut);
  k_squash<<<(BB * PC + 255) / 256, 256, 0, stream>>>(ut, unorm);

  // capsule transform (streams 268 MB of W)
  k_uhat<<<dim3(PC, LL), 64, 0, stream>>>(W, ut, unorm, uhat);

  // dynamic routing: 3 iterations
  for (int it = 0; it < 3; ++it) {
    k_softmax<<<(BB * PC + 255) / 256, 256, 0, stream>>>(blog, cbuf);
    k_sv<<<BB * LL, 256, 0, stream>>>(cbuf, uhat, vbuf);
    if (it < 2) {
      k_zero<<<1, 32, 0, stream>>>(dsum, 1);
      k_dist<<<BB * LL, 256, 0, stream>>>(vbuf, uhat, dbuf, dsum);
      k_blog<<<(BB * LL * PC + 255) / 256, 256, 0, stream>>>(dbuf, dsum, blog);
    }
  }

  k_T<<<1, 256, 0, stream>>>(cbuf, T_o);
  k_vjD<<<1, 256, 0, stream>>>(vbuf, vj_o, D_o);

  // decoder: one workgroup per primitive, ~137KB dynamic LDS (<=320KB/WGP)
  size_t smem = (size_t)(2 * BOT * 256 + 256) * sizeof(float);
  (void)hipFuncSetAttribute(reinterpret_cast<const void*>(k_decoder),
                            hipFuncAttributeMaxDynamicSharedMemorySize, (int)smem);
  k_decoder<<<NPRIM, 256, smem, stream>>>(rg, vbuf,
      d1w, d1b, d1g, d1be, d2w, d2b, d2g, d2be,
      d3w, d3b, d3g, d3be, d4w, d4b, rc_o);
  (void)in_sizes; (void)n_in; (void)out_size; (void)ws_size;
}
